// SelfAttention_82394652606635
// MI455X (gfx1250) — matrix operand order repaired and checked by reading, not compiled, not run on hardware
//
#include <hip/hip_runtime.h>
#include <hip/hip_bf16.h>
#include <stdint.h>

// ---------------------------------------------------------------------------
// Sizes (fixed by the reference)
// ---------------------------------------------------------------------------
#define EMBED 1024
#define NHEAD 16
#define HDIM  64
#define BATCH 4
#define SEQ   1024
#define NTOK  (BATCH * SEQ)     // 4096
#define EPSLN 1e-5f
#define NEGV  (-1e9f)

typedef __attribute__((ext_vector_type(16))) __bf16 v16bf;
typedef __attribute__((ext_vector_type(8)))  __bf16 v8bf;
typedef __attribute__((ext_vector_type(8)))  float  v8f;
typedef __attribute__((ext_vector_type(4)))  unsigned int u32x4;
typedef __attribute__((ext_vector_type(4)))  int i32x4;
typedef __attribute__((ext_vector_type(8)))  int i32x8;

static __device__ __forceinline__ v16bf cat8(v8bf lo, v8bf hi) {
    return __builtin_shufflevector(lo, hi, 0,1,2,3,4,5,6,7,8,9,10,11,12,13,14,15);
}
static __device__ __forceinline__ v8bf ldv8(const __bf16* p) {
    return *(const v8bf*)p;   // 16-byte aligned by construction
}

// ---------------------------------------------------------------------------
// TDM: 2-D bf16 tile load  global -> LDS  (D# per CDNA5 ISA)
//   tile_d0 = contiguous dim (elements), tile_d1 = rows; row stride in elements.
// ---------------------------------------------------------------------------
static __device__ __forceinline__ void tdm_load_2d_bf16(
        unsigned lds_byte, unsigned long long gaddr,
        unsigned tile_d0, unsigned tile_d1,
        unsigned tens_d0, unsigned tens_d1, unsigned stride_el) {
    u32x4 g0;
    g0.x = 1u;                                         // count=1, is_restore=0, no gather
    g0.y = lds_byte;                                   // lds_addr [63:32]
    g0.z = (unsigned)gaddr;                            // global_addr [95:64]
    g0.w = (unsigned)((gaddr >> 32) & 0x1FFFFFFu)      // global_addr [120:96]
         | (2u << 30);                                 // type = 2 ("image")
    i32x8 g1;
    g1[0] = (int)(1u << 16);                           // data_size = 1 -> 2 bytes
    g1[1] = (int)((tens_d0 & 0xFFFFu) << 16);          // tensor_dim0[15:0]
    g1[2] = (int)(((tens_d0 >> 16) & 0xFFFFu)          // tensor_dim0[31:16]
         | ((tens_d1 & 0xFFFFu) << 16));               // tensor_dim1[15:0]
    g1[3] = (int)(((tens_d1 >> 16) & 0xFFFFu)          // tensor_dim1[31:16]
         | ((tile_d0 & 0xFFFFu) << 16));               // tile_dim0
    g1[4] = (int)(tile_d1 & 0xFFFFu);                  // tile_dim1 (tile_dim2 = 0)
    g1[5] = (int)stride_el;                            // tensor_dim0_stride[31:0]
    g1[6] = 0;                                         // stride hi / dim1_stride lo
    g1[7] = 0;
    i32x4 z4 = (i32x4){0, 0, 0, 0};
#if __has_include(<hip/amd_detail/amd_gfx1250_TDM.h>)
    i32x8 z8 = (i32x8){0, 0, 0, 0, 0, 0, 0, 0};
    __builtin_amdgcn_tensor_load_to_lds(g0, g1, z4, z4, z8, 0);
#else
    __builtin_amdgcn_tensor_load_to_lds(g0, g1, z4, z4, 0);
#endif
}

// ---------------------------------------------------------------------------
// Kernel 0a: fp32 -> bf16 elementwise convert (x)
// ---------------------------------------------------------------------------
__global__ __launch_bounds__(256) void k_convert(const float* __restrict__ in,
                                                 __bf16* __restrict__ out, int n) {
    for (int i = blockIdx.x * blockDim.x + threadIdx.x; i < n; i += gridDim.x * blockDim.x)
        out[i] = (__bf16)in[i];
}

// ---------------------------------------------------------------------------
// Kernel 0b: W [K=1024][N=1024] fp32  ->  Wt [N][K] bf16 (LDS-tiled transpose)
// ---------------------------------------------------------------------------
__global__ __launch_bounds__(256) void k_transpose(const float* __restrict__ W,
                                                   __bf16* __restrict__ Wt) {
    __shared__ float tile[32][33];
    int x  = blockIdx.x * 32 + threadIdx.x;      // n
    int y0 = blockIdx.y * 32;                    // k base
    #pragma unroll
    for (int j = threadIdx.y; j < 32; j += 8)
        tile[j][threadIdx.x] = W[(size_t)(y0 + j) * EMBED + x];
    __syncthreads();
    int xo  = blockIdx.y * 32 + threadIdx.x;     // k
    int yo0 = blockIdx.x * 32;                   // n base
    #pragma unroll
    for (int j = threadIdx.y; j < 32; j += 8)
        Wt[(size_t)(yo0 + j) * EMBED + xo] = (__bf16)tile[threadIdx.x][j];
}

// ---------------------------------------------------------------------------
// Kernel 1: P = x @ W + bias -> LayerNorm -> ReLU -> bf16
//   block = 256 thr (8 waves), one block owns 16 full rows (all 1024 cols)
//   B (weight) panels [1024 n x 32 k] staged via TDM into double-buffered LDS
// ---------------------------------------------------------------------------
__global__ __launch_bounds__(256) void k_proj_ln_relu(
        const __bf16* __restrict__ xb,    // [NTOK][EMBED]
        const __bf16* __restrict__ Wt,    // [EMBED(n)][EMBED(k)]  (transposed)
        const float*  __restrict__ bias,
        const float*  __restrict__ gamma,
        const float*  __restrict__ beta,
        __bf16* __restrict__ outb) {      // [NTOK][EMBED]
    extern __shared__ char psmem[];
    __bf16* sB0   = (__bf16*)(psmem);              // 64 KB : 1024 cols x 32 k
    __bf16* sB1   = (__bf16*)(psmem + 65536);      // 64 KB
    float*  s_sum = (float*)(psmem + 131072);      // 16
    float*  s_sq  = s_sum + 16;                    // 16

    const int tid  = threadIdx.x;
    const int lane = tid & 31;
    const int w    = tid >> 5;
    const int r0   = blockIdx.x * 16;
    const int m16  = lane & 15;
    const int hi   = (lane & 16) ? 1 : 0;
    const int ao   = hi ? 8  : 0;        // A operand K-chunk select
    const int bo   = hi ? 16 : 0;        // unused; B operand uses ao like A

    v8f acc[8];
    #pragma unroll
    for (int t = 0; t < 8; ++t) acc[t] = (v8f){};

    const __bf16* xrow = xb + (size_t)(r0 + m16) * EMBED;
    __bf16* sB[2] = { sB0, sB1 };
    const unsigned long long wbase = (unsigned long long)(uintptr_t)Wt;

    // prologue: DMA first weight panel (k = 0..31, all 1024 columns)
    if (w == 0)
        tdm_load_2d_bf16((unsigned)(uintptr_t)sB0, wbase,
                         /*tile*/32, 1024, /*tensor*/EMBED, EMBED, EMBED);

    for (int i = 0; i < EMBED / 32; ++i) {
        const int kb = i * 32;
        if (w == 0) {
            if (i + 1 < EMBED / 32) {
                tdm_load_2d_bf16((unsigned)(uintptr_t)sB[(i + 1) & 1],
                                 wbase + (unsigned long long)(kb + 32) * 2,
                                 32, 1024, EMBED, EMBED, EMBED);
                __builtin_amdgcn_s_wait_tensorcnt(1);  // current panel complete
            } else {
                __builtin_amdgcn_s_wait_tensorcnt(0);
            }
        }
        __syncthreads();                               // publish panel to all waves

        const __bf16* Bb = sB[i & 1];
        v16bf a = cat8(ldv8(xrow + kb + ao), ldv8(xrow + kb + 16 + ao));
        #pragma unroll
        for (int t = 0; t < 8; ++t) {
            const int col = w * 128 + t * 16 + m16;
            const __bf16* br = Bb + col * 32 + ao;     // LDS: ds_load_b128 x2
            v16bf bm = cat8(ldv8(br), ldv8(br + 16));
            acc[t] = __builtin_amdgcn_wmma_f32_16x16x32_bf16(
                         false, a, false, bm, (short)0, acc[t], false, false);
        }
        __syncthreads();                               // reads done before next DMA reuses buf
    }

    // add bias (col-constant per tile)
    #pragma unroll
    for (int t = 0; t < 8; ++t) {
        const float bv = bias[w * 128 + t * 16 + m16];
        #pragma unroll
        for (int j = 0; j < 8; ++j) acc[t][j] += bv;
    }

    if (tid < 16) { s_sum[tid] = 0.f; s_sq[tid] = 0.f; }
    __syncthreads();

    // row statistics across the whole block via LDS float atomics (ds_add_f32)
    #pragma unroll
    for (int j = 0; j < 8; ++j) {
        float p = 0.f, q = 0.f;
        #pragma unroll
        for (int t = 0; t < 8; ++t) { float v = acc[t][j]; p += v; q += v * v; }
        const int row = hi ? (j + 8) : j;
        atomicAdd(&s_sum[row], p);
        atomicAdd(&s_sq[row],  q);
    }
    __syncthreads();

    // normalize + affine + ReLU, store bf16
    #pragma unroll
    for (int j = 0; j < 8; ++j) {
        const int   row = hi ? (j + 8) : j;
        const float mu  = s_sum[row] * (1.0f / EMBED);
        const float var = s_sq[row] * (1.0f / EMBED) - mu * mu;
        const float rin = rsqrtf(var + EPSLN);
        #pragma unroll
        for (int t = 0; t < 8; ++t) {
            const int col = w * 128 + t * 16 + m16;
            float y = (acc[t][j] - mu) * rin * gamma[col] + beta[col];
            y = fmaxf(y, 0.f);
            outb[(size_t)(r0 + row) * EMBED + col] = (__bf16)y;
        }
    }
}

// ---------------------------------------------------------------------------
// Kernel 2: scores -> symmetric mask -> softmax -> mean over heads
//   grid = (SEQ/16, BATCH); block = 256 thr (8 waves)
//   dynamic LDS: s_sc[16*1024] + s_acc[16*1024] + reductions
// ---------------------------------------------------------------------------
__global__ __launch_bounds__(256) void k_attn(
        const __bf16* __restrict__ Qb,    // [NTOK][EMBED]
        const __bf16* __restrict__ Kb,    // [NTOK][EMBED]
        const unsigned char* __restrict__ mask, // [B][S][S] bool
        float* __restrict__ out) {        // [B][S][S]
    extern __shared__ float smem[];
    float* s_sc  = smem;                  // 16*1024
    float* s_acc = smem + 16 * 1024;      // 16*1024
    float* s_red = smem + 32 * 1024;      // 256
    float* s_max = s_red + 256;           // 16
    float* s_sum = s_max + 16;            // 16

    const int tid  = threadIdx.x;
    const int lane = tid & 31;
    const int w    = tid >> 5;
    const int b    = blockIdx.y;
    const int q0   = blockIdx.x * 16;
    const int m16  = lane & 15;
    const int hi   = (lane & 16) ? 1 : 0;
    const int ao   = hi ? 8  : 0;
    const int bo   = hi ? 16 : 0;
    const size_t tokbase = (size_t)b * SEQ;

    for (int i = tid; i < 16 * 1024; i += 256) s_acc[i] = 0.f;
    __syncthreads();

    const __bf16* qrow = Qb + (tokbase + q0 + m16) * EMBED;

    // softmax-phase thread mapping: 16 threads per query row, 64 keys each
    const int row = tid & 15;
    const int c0  = (tid >> 4) * 64;
    const unsigned char* mrow = mask + (size_t)b * SEQ * SEQ + (size_t)(q0 + row) * SEQ;
    const unsigned char* mcol = mask + (size_t)b * SEQ * SEQ + (q0 + row);

    for (int h = 0; h < NHEAD; ++h) {
        const int hb = h * HDIM;
        if (h + 1 < NHEAD)                       // global_prefetch_b8: next head's Q slice
            __builtin_prefetch(qrow + hb + HDIM, 0, 1);

        v8f acc[8];
        #pragma unroll
        for (int t = 0; t < 8; ++t) acc[t] = (v8f){};

        #pragma unroll
        for (int kb = 0; kb < HDIM; kb += 32) {
            v16bf a = cat8(ldv8(qrow + hb + kb + ao), ldv8(qrow + hb + kb + 16 + ao));
            #pragma unroll
            for (int t = 0; t < 8; ++t) {
                const int key = w * 128 + t * 16 + m16;
                const __bf16* kr = Kb + (tokbase + key) * EMBED + hb + kb + ao;
                v16bf bm = cat8(ldv8(kr), ldv8(kr + 16));
                acc[t] = __builtin_amdgcn_wmma_f32_16x16x32_bf16(
                             false, a, false, bm, (short)0, acc[t], false, false);
            }
        }

        // scatter scaled scores to LDS (C layout: vgpr j -> row j / j+8)
        #pragma unroll
        for (int t = 0; t < 8; ++t) {
            const int kc = w * 128 + t * 16 + m16;
            #pragma unroll
            for (int j = 0; j < 8; ++j) {
                const int lr = hi ? (j + 8) : j;
                s_sc[lr * 1024 + kc] = acc[t][j] * 0.125f;   // 1/sqrt(64)
            }
        }
        __syncthreads();

        // mask + row max
        float mx = -3.4e38f;
        #pragma unroll 4
        for (int i = 0; i < 64; ++i) {
            const int k = c0 + i;
            const bool msk = (mrow[k] | mcol[(size_t)k * SEQ]) != 0;
            float v = msk ? NEGV : s_sc[row * 1024 + k];
            s_sc[row * 1024 + k] = v;
            mx = fmaxf(mx, v);
        }
        s_red[row * 16 + (tid >> 4)] = mx;
        __syncthreads();
        if (tid < 16) {
            float m = -3.4e38f;
            #pragma unroll
            for (int i = 0; i < 16; ++i) m = fmaxf(m, s_red[tid * 16 + i]);
            s_max[tid] = m;
        }
        __syncthreads();

        // exp + row sum
        const float rmax = s_max[row];
        float sm = 0.f;
        #pragma unroll 4
        for (int i = 0; i < 64; ++i) {
            float e = __expf(s_sc[row * 1024 + c0 + i] - rmax);
            s_sc[row * 1024 + c0 + i] = e;
            sm += e;
        }
        s_red[row * 16 + (tid >> 4)] = sm;
        __syncthreads();
        if (tid < 16) {
            float s = 0.f;
            #pragma unroll
            for (int i = 0; i < 16; ++i) s += s_red[tid * 16 + i];
            s_sum[tid] = s;
        }
        __syncthreads();

        // accumulate head mean
        const float rinv = 1.0f / s_sum[row];
        #pragma unroll 4
        for (int i = 0; i < 64; ++i)
            s_acc[row * 1024 + c0 + i] += s_sc[row * 1024 + c0 + i] * rinv;
        __syncthreads();   // s_sc reused next head
    }

    // write [B,S,S] output
    float* orow = out + (size_t)b * SEQ * SEQ + (size_t)(q0 + row) * SEQ;
    #pragma unroll 4
    for (int i = 0; i < 64; ++i)
        orow[c0 + i] = s_acc[row * 1024 + c0 + i] * (1.0f / NHEAD);
}

// ---------------------------------------------------------------------------
// Host launcher
// ---------------------------------------------------------------------------
extern "C" void kernel_launch(void* const* d_in, const int* in_sizes, int n_in,
                              void* d_out, int out_size, void* d_ws, size_t ws_size,
                              hipStream_t stream) {
    const float* x    = (const float*)d_in[0];
    const unsigned char* mask = (const unsigned char*)d_in[1];
    const float* Wq   = (const float*)d_in[2];
    const float* bq   = (const float*)d_in[3];
    const float* Wk   = (const float*)d_in[4];
    const float* bk   = (const float*)d_in[5];
    const float* gq   = (const float*)d_in[6];
    const float* bq_n = (const float*)d_in[7];
    const float* gk   = (const float*)d_in[8];
    const float* bk_n = (const float*)d_in[9];
    float* out = (float*)d_out;

    // workspace layout (bf16 staging), total 28 MB
    char* wsb = (char*)d_ws;
    __bf16* xb  = (__bf16*)(wsb);                          // 8 MB
    __bf16* Wqt = (__bf16*)(wsb + (size_t)8  * (1 << 20)); // 2 MB
    __bf16* Wkt = (__bf16*)(wsb + (size_t)10 * (1 << 20)); // 2 MB
    __bf16* Qb  = (__bf16*)(wsb + (size_t)12 * (1 << 20)); // 8 MB
    __bf16* Kb  = (__bf16*)(wsb + (size_t)20 * (1 << 20)); // 8 MB

    // 0) precision staging
    k_convert<<<2048, 256, 0, stream>>>(x, xb, NTOK * EMBED);
    k_transpose<<<dim3(32, 32), dim3(32, 8), 0, stream>>>(Wq, Wqt);
    k_transpose<<<dim3(32, 32), dim3(32, 8), 0, stream>>>(Wk, Wkt);

    // 1) fused projection + LayerNorm + ReLU (Q then K), TDM-staged weights
    const size_t proj_smem = 2 * 65536 + 32 * sizeof(float);   // 128 KB + stats
    k_proj_ln_relu<<<NTOK / 16, 256, proj_smem, stream>>>(xb, Wqt, bq, gq, bq_n, Qb);
    k_proj_ln_relu<<<NTOK / 16, 256, proj_smem, stream>>>(xb, Wkt, bk, gk, bk_n, Kb);

    // 2) fused scores + mask + softmax + head-mean (scores never hit HBM)
    const size_t smem_bytes = (size_t)(32 * 1024 + 256 + 32) * sizeof(float); // 132 KB
    k_attn<<<dim3(SEQ / 16, BATCH), 256, smem_bytes, stream>>>(Qb, Kb, mask, out);
}
